// GgmlDecoderAttention_21449066676480
// MI455X (gfx1250) — matrix operand order repaired and checked by reading, not compiled, not run on hardware
//
#include <hip/hip_runtime.h>
#include <hip/hip_bf16.h>
#include <stdint.h>

// ---------------------------------------------------------------------------
// Types
// ---------------------------------------------------------------------------
typedef __attribute__((ext_vector_type(16))) __bf16       v16bf;
typedef __attribute__((ext_vector_type(8)))  float        v8f;
typedef __attribute__((ext_vector_type(4)))  unsigned int v4u;
typedef __attribute__((ext_vector_type(8)))  int          v8i;
typedef __attribute__((ext_vector_type(4)))  int          v4i;

// ---------------------------------------------------------------------------
// Problem constants (static per reference)
// ---------------------------------------------------------------------------
#define BATCH   16
#define TQ      16
#define DIMW    4096
#define HQ      32
#define HKV     8
#define NREP    4
#define HD      128
#define SEQ     8192
#define NSPLIT  8
#define SBLK    32            // S columns per inner block
#define KVPITCH 132           // padded LDS row pitch (floats) for K/V tiles

// ---------------------------------------------------------------------------
// WMMA helpers (wave32; A/B use contiguous k order k = 16*(lane/16) + s).
// All fragment loads are 32-bit-word based to avoid 16-bit DS traffic and
// v_mov_b16 packing.
// ---------------------------------------------------------------------------
__device__ inline v8f vzero8() {
  v8f z;
#pragma unroll
  for (int i = 0; i < 8; ++i) z[i] = 0.0f;
  return z;
}

__device__ inline v8f wmma_bf16(v16bf a, v16bf b, v8f c) {
  return __builtin_amdgcn_wmma_f32_16x16x32_bf16(false, a, false, b, (short)0, c,
                                                 false, false);
}

__device__ inline unsigned int pack_bf16(float lo, float hi) {
  union { __bf16 b[2]; unsigned int u; } r;
  r.b[0] = (__bf16)lo;
  r.b[1] = (__bf16)hi;
  return r.u;
}

// A (16x32 bf16, row-major src, stride ld in elements; ld must be even).
// lanes 0-15: row=lane, word p -> K-pair = 2*p
// lanes 16-31: same rows, K offset +16
__device__ inline v16bf loadA_bf16(const __bf16* src, int ld) {
  union { v16bf v; unsigned int u[8]; } r;
  const int lane = threadIdx.x & 31;
  const int row  = lane & 15;
  const int koff = (lane >> 4) * 16;
  const unsigned int* s32 = (const unsigned int*)(src + row * ld);
#pragma unroll
  for (int p = 0; p < 8; ++p) {
    const int k = koff + (p << 1);
    r.u[p] = s32[k >> 1];
  }
  return r.v;
}

// B (32x16) from an LDS image pre-packed as uint pairs over K:
// sB2[kk][n] holds {B[2kk][n], B[2kk+1][n]}; ld2 = row pitch in uints.
// lanes 0-15: n=lane, kk = p ; lanes 16-31: n=lane-16, kk = 8+p
__device__ inline v16bf loadB_pk(const unsigned int* sB2, int ld2) {
  union { v16bf v; unsigned int u[8]; } r;
  const int lane = threadIdx.x & 31;
  const int n    = lane & 15;
  const int kkb  = (lane >> 4) * 8;
#pragma unroll
  for (int p = 0; p < 8; ++p) r.u[p] = sB2[(kkb + p) * ld2 + n];
  return r.v;
}

// B (32x16) built from a transposed f32 source: B[k][n] = src[n*ld + k].
// Pairs over k are contiguous -> float2 read + pack.
__device__ inline v16bf loadB_tr_f32(const float* src, int ld) {
  union { v16bf v; unsigned int u[8]; } r;
  const int lane = threadIdx.x & 31;
  const int n    = lane & 15;
  const int kb   = (lane >> 4) * 16;
#pragma unroll
  for (int p = 0; p < 8; ++p) {
    const float2 w = *(const float2*)(src + n * ld + kb + 2 * p);
    r.u[p] = pack_bf16(w.x, w.y);
  }
  return r.v;
}

// B (32x16) from row-major f32 K x N source (pairs stride ld apart).
__device__ inline v16bf loadB_rm_f32(const float* src, int ld) {
  union { v16bf v; unsigned int u[8]; } r;
  const int lane = threadIdx.x & 31;
  const int n    = lane & 15;
  const int kb   = (lane >> 4) * 16;
#pragma unroll
  for (int p = 0; p < 8; ++p)
    r.u[p] = pack_bf16(src[(kb + 2 * p) * ld + n], src[(kb + 2 * p + 1) * ld + n]);
  return r.v;
}

__device__ inline float hmax16(float v) {
#pragma unroll
  for (int m = 1; m < 16; m <<= 1) v = fmaxf(v, __shfl_xor(v, m, 32));
  return v;
}
__device__ inline float hsum16(float v) {
#pragma unroll
  for (int m = 1; m < 16; m <<= 1) v += __shfl_xor(v, m, 32);
  return v;
}

// ---------------------------------------------------------------------------
// Tensor Data Mover: 2D tile load, 32 rows x 128 f32, src row stride 1024 f32,
// LDS padded +4 dwords per 128-dword row (pitch 132 floats).
// D# layout per CDNA5 ISA.
// ---------------------------------------------------------------------------
__device__ inline void tdm_load_kv_tile(unsigned int lds_byte,
                                        unsigned long long gaddr) {
  v4u g0;
  g0[0] = 1u;                                   // count=1, no gather
  g0[1] = lds_byte;                             // LDS address (bytes)
  g0[2] = (unsigned int)gaddr;                  // global addr low 32
  g0[3] = (unsigned int)((gaddr >> 32) & 0x1FFFFFFull) | (2u << 30); // type=2
  v8i g1;
  // data_size=4B (2<<16) | pad_enable(1<<20) | pad_interval=6 (128 dw, <<22)
  // | pad_amount=3 (4 dw, <<25)
  g1[0] = (int)((2u << 16) | (1u << 20) | (6u << 22) | (3u << 25));
  g1[1] = (int)(128u << 16);                    // tensor_dim0 = 128 (lo16)
  g1[2] = (int)(32u << 16);                     // dim0 hi=0 | tensor_dim1 = 32
  g1[3] = (int)(128u << 16);                    // dim1 hi=0 | tile_dim0 = 128
  g1[4] = (int)(32u);                           // tile_dim1 = 32, tile_dim2 = 0
  g1[5] = (int)(HKV * HD);                      // tensor_dim0_stride = 1024
  g1[6] = 0;
  g1[7] = 0;
  v4i gz = {0, 0, 0, 0};
#if defined(__clang_major__) && (__clang_major__ >= 23)
  v8i gz8 = {0, 0, 0, 0, 0, 0, 0, 0};
  __builtin_amdgcn_tensor_load_to_lds(g0, g1, gz, gz, gz8, 0);
#else
  __builtin_amdgcn_tensor_load_to_lds(g0, g1, gz, gz, 0);
#endif
}

// ---------------------------------------------------------------------------
// Kernel 1/4: C[M,N] = A[M,K] * W[N,K]^T   (f32 in/out, bf16 WMMA compute)
// Block: 256 threads (8 waves), tile 128x128, waves 2(M)x4(N) -> 64x32 each.
// B tile staged in LDS as packed bf16 pairs (uint) so fragment loads are b32.
// ---------------------------------------------------------------------------
__global__ __launch_bounds__(256) void gemm_xwT(const float* __restrict__ A,
                                                const float* __restrict__ W,
                                                float* __restrict__ C,
                                                int M, int N, int K) {
  __shared__ __bf16       sA[128][34];
  __shared__ unsigned int sB2[16][132];  // [k/2][n] packed pairs
  const int m0   = blockIdx.y * 128;
  const int n0   = blockIdx.x * 128;
  const int tid  = threadIdx.x;
  const int wave = tid >> 5;
  const int mw   = (wave >> 2) * 64;
  const int nw   = (wave & 3) * 32;

  v8f acc[4][2];
#pragma unroll
  for (int i = 0; i < 4; ++i)
#pragma unroll
    for (int j = 0; j < 2; ++j) acc[i][j] = vzero8();

  for (int k0 = 0; k0 < K; k0 += 32) {
    {  // stage A tile (128x32) as bf16
      const int row = tid >> 1;
      const int cb  = (tid & 1) * 16;
      const float* src = A + (size_t)(m0 + row) * K + k0 + cb;
#pragma unroll
      for (int i = 0; i < 16; ++i) sA[row][cb + i] = (__bf16)src[i];
    }
    {  // stage W tile transposed + pair-packed: sB2[k/2][n]
      const int n   = tid >> 1;
      const int kkb = (tid & 1) * 8;
      const float* src = W + (size_t)(n0 + n) * K + k0 + kkb * 2;
#pragma unroll
      for (int i = 0; i < 8; ++i)
        sB2[kkb + i][n] = pack_bf16(src[2 * i], src[2 * i + 1]);
    }
    __syncthreads();

    v16bf bfz[2];
#pragma unroll
    for (int j = 0; j < 2; ++j) bfz[j] = loadB_pk(&sB2[0][nw + 16 * j], 132);
#pragma unroll
    for (int i = 0; i < 4; ++i) {
      v16bf af = loadA_bf16(&sA[mw + 16 * i][0], 34);
#pragma unroll
      for (int j = 0; j < 2; ++j) acc[i][j] = wmma_bf16(af, bfz[j], acc[i][j]);
    }
    __syncthreads();
  }

  const int lane = tid & 31;
  const int cn   = lane & 15;
  const int cm   = (lane >> 4) * 8;
#pragma unroll
  for (int i = 0; i < 4; ++i)
#pragma unroll
    for (int j = 0; j < 2; ++j)
#pragma unroll
      for (int p = 0; p < 8; ++p)
        C[(size_t)(m0 + mw + 16 * i + cm + p) * N + (n0 + nw + 16 * j + cn)] =
            acc[i][j][p];
}

// ---------------------------------------------------------------------------
// Kernel 2: RoPE q/k + scatter to attention-friendly layouts.
//   q_att  [B][HQ][T][HD]  bf16 (roped)
//   k_new  [B][HKV][T][HD] f32  (roped)    -- the 16 "cache tail" rows
//   v_new  [B][HKV][T][HD] f32
// ---------------------------------------------------------------------------
__global__ void rope_scatter(const float* __restrict__ qraw,
                             const float* __restrict__ kraw,
                             const float* __restrict__ vraw,
                             const int* __restrict__ input_pos,
                             __bf16* __restrict__ q_att,
                             float* __restrict__ k_new,
                             float* __restrict__ v_new) {
  const int gid = blockIdx.x * blockDim.x + threadIdx.x;
  const int QP = BATCH * TQ * HQ * (HD / 2);
  const int KP = BATCH * TQ * HKV * (HD / 2);
  const int VP = KP;
  const float LNB_OVER_HD = 13.122363377404328f / (float)HD;  // ln(5e5)/128

  if (gid < QP) {
    const int i  = gid & 63;
    const int h  = (gid >> 6) & 31;
    const int bt = gid >> 11;
    const int t = bt & 15, b = bt >> 4;
    const float x0 = qraw[(size_t)bt * DIMW + h * HD + 2 * i];
    const float x1 = qraw[(size_t)bt * DIMW + h * HD + 2 * i + 1];
    const float pos = (float)input_pos[t];
    const float ang = pos * __expf(-(float)(2 * i) * LNB_OVER_HD);
    float s, c;
    __sincosf(ang, &s, &c);
    const size_t o = ((size_t)(b * HQ + h) * TQ + t) * HD + 2 * i;
    q_att[o]     = (__bf16)(x0 * c - x1 * s);
    q_att[o + 1] = (__bf16)(x0 * s + x1 * c);
  } else if (gid < QP + KP) {
    const int g2 = gid - QP;
    const int i  = g2 & 63;
    const int g  = (g2 >> 6) & 7;
    const int bt = g2 >> 9;
    const int t = bt & 15, b = bt >> 4;
    const float x0 = kraw[(size_t)bt * (HKV * HD) + g * HD + 2 * i];
    const float x1 = kraw[(size_t)bt * (HKV * HD) + g * HD + 2 * i + 1];
    const float pos = (float)input_pos[t];
    const float ang = pos * __expf(-(float)(2 * i) * LNB_OVER_HD);
    float s, c;
    __sincosf(ang, &s, &c);
    const size_t o = ((size_t)(b * HKV + g) * TQ + t) * HD + 2 * i;
    k_new[o]     = x0 * c - x1 * s;
    k_new[o + 1] = x0 * s + x1 * c;
  } else if (gid < QP + KP + VP) {
    const int g2 = gid - QP - KP;
    const int i  = g2 & 63;
    const int g  = (g2 >> 6) & 7;
    const int bt = g2 >> 9;
    const int t = bt & 15, b = bt >> 4;
    const size_t si = (size_t)bt * (HKV * HD) + g * HD + 2 * i;
    const size_t o  = ((size_t)(b * HKV + g) * TQ + t) * HD + 2 * i;
    v_new[o]     = vraw[si];
    v_new[o + 1] = vraw[si + 1];
  }
}

// ---------------------------------------------------------------------------
// Kernel 3: flash-decode attention, split over S.
// Grid (b, g, split); block = 128 threads = 4 waves; wave r handles q-head
// h = g*4+r (16 query rows). K/V tiles streamed by TDM into double-buffered
// padded LDS; the next tile is issued before computing the current one, and
// the issuing waves drain with s_wait_tensorcnt(1).
// ---------------------------------------------------------------------------
__global__ __launch_bounds__(128) void attn_kernel(
    const float* __restrict__ k_cache, const float* __restrict__ v_cache,
    const __bf16* __restrict__ q_att, const float* __restrict__ k_new,
    const float* __restrict__ v_new, float* __restrict__ y_part,
    float* __restrict__ m_part, float* __restrict__ l_part) {
  __shared__ float  sK[2][SBLK][KVPITCH];
  __shared__ float  sV[2][SBLK][KVPITCH];
  __shared__ __bf16 sP[4][16][34];

  const int b = blockIdx.x, g = blockIdx.y, split = blockIdx.z;
  const int tid = threadIdx.x, wave = tid >> 5, lane = tid & 31;
  const int r = wave;
  const int h = g * NREP + r;
  const float scale = 0.08838834764831845f;  // 1/sqrt(128)

  // Q A-fragments: 16 rows x 128 dims -> 4 fragments of K=32 (kept in regs)
  const __bf16* qbase = q_att + ((size_t)(b * HQ + h) * TQ) * HD;
  v16bf qf[4];
#pragma unroll
  for (int c = 0; c < 4; ++c) qf[c] = loadA_bf16(qbase + 32 * c, HD);

  v8f acc[8];
#pragma unroll
  for (int n = 0; n < 8; ++n) acc[n] = vzero8();
  float m8[8], l8[8];
#pragma unroll
  for (int p = 0; p < 8; ++p) { m8[p] = -1e30f; l8[p] = 0.0f; }

  const int s_begin = split * 1024;  // splits 0..6: 1024 cols; split 7: 1008+16
  const int nblk = 32;
  const int mrow = (lane >> 4) * 8;
  const int cn   = lane & 15;
  const unsigned long long kvoff =
      ((unsigned long long)b * SEQ * HKV + g) * HD * 4ull;  // + s*HKV*HD*4

  // ---- prologue: issue tile 0 -------------------------------------------
  int cur = 0;
  {
    const unsigned long long so = (unsigned long long)s_begin * HKV * HD * 4ull;
    if (wave == 0)
      tdm_load_kv_tile((unsigned int)(uintptr_t)&sK[0][0][0],
                       (unsigned long long)(uintptr_t)k_cache + kvoff + so);
    else if (wave == 1)
      tdm_load_kv_tile((unsigned int)(uintptr_t)&sV[0][0][0],
                       (unsigned long long)(uintptr_t)v_cache + kvoff + so);
  }

  for (int ib = 0; ib < nblk; ++ib) {
    const bool mixed      = (split == NSPLIT - 1) && (ib == nblk - 1);
    const bool next_mixed = (split == NSPLIT - 1) && (ib + 1 == nblk - 1);
    const bool issue_next = (ib + 1 < nblk) && !next_mixed;
    const int  nxt = cur ^ 1;

    if (issue_next) {  // prefetch tile ib+1 into the other buffer
      const unsigned long long so =
          (unsigned long long)(s_begin + (ib + 1) * SBLK) * HKV * HD * 4ull;
      if (wave == 0)
        tdm_load_kv_tile((unsigned int)(uintptr_t)&sK[nxt][0][0],
                         (unsigned long long)(uintptr_t)k_cache + kvoff + so);
      else if (wave == 1)
        tdm_load_kv_tile((unsigned int)(uintptr_t)&sV[nxt][0][0],
                         (unsigned long long)(uintptr_t)v_cache + kvoff + so);
    }

    if (!mixed) {
      // drain current tile on the issuing waves (1 outstanding = the prefetch)
      if (wave < 2) {
        if (issue_next) __builtin_amdgcn_s_wait_tensorcnt(1);
        else            __builtin_amdgcn_s_wait_tensorcnt(0);
      }
    } else {
      // tail block: rows 0-15 = cache s=8160..8175, rows 16-31 = new K/V
      for (int e = tid; e < SBLK * (HD / 4); e += 128) {
        const int row = e >> 5;  // HD/4 = 32 float4 per row
        const int c4  = e & 31;
        const float4* srck;
        const float4* srcv;
        if (row < 16) {
          const size_t o = (((size_t)b * SEQ + 8160 + row) * HKV + g) * HD;
          srck = (const float4*)(k_cache + o) + c4;
          srcv = (const float4*)(v_cache + o) + c4;
        } else {
          const size_t o = ((size_t)(b * HKV + g) * TQ + (row - 16)) * HD;
          srck = (const float4*)(k_new + o) + c4;
          srcv = (const float4*)(v_new + o) + c4;
        }
        *((float4*)&sK[cur][row][0] + c4) = *srck;
        *((float4*)&sV[cur][row][0] + c4) = *srcv;
      }
    }
    __syncthreads();  // publish current tile to all waves

    const float(*Kt)[KVPITCH] = sK[cur];
    const float(*Vt)[KVPITCH] = sV[cur];

    // ---- scores: 2 tiles of 16x16 over this 32-column block -------------
    v8f st[2];
#pragma unroll
    for (int nc = 0; nc < 2; ++nc) {
      v8f s = vzero8();
#pragma unroll
      for (int c = 0; c < 4; ++c) {
        v16bf bk = loadB_tr_f32(&Kt[nc * 16][c * 32], KVPITCH);
        s = wmma_bf16(qf[c], bk, s);
      }
#pragma unroll
      for (int p = 0; p < 8; ++p) s[p] *= scale;
      st[nc] = s;
    }
    if (mixed) {
      // columns 16..31 are new positions j=0..15; allowed iff j <= t(row)
#pragma unroll
      for (int p = 0; p < 8; ++p)
        if (cn > mrow + p) st[1][p] = -1e30f;
    }

    // ---- online softmax --------------------------------------------------
    float scl[8], rs[8];
#pragma unroll
    for (int p = 0; p < 8; ++p) {
      float bm = hmax16(fmaxf(st[0][p], st[1][p]));
      const float mn = fmaxf(m8[p], bm);
      scl[p] = __expf(m8[p] - mn);
      m8[p]  = mn;
    }
#pragma unroll
    for (int p = 0; p < 8; ++p) {
      const float e0 = __expf(st[0][p] - m8[p]);
      const float e1 = __expf(st[1][p] - m8[p]);
      sP[r][mrow + p][cn]      = (__bf16)e0;
      sP[r][mrow + p][16 + cn] = (__bf16)e1;
      rs[p] = e0 + e1;
    }
#pragma unroll
    for (int p = 0; p < 8; ++p) {
      rs[p] = hsum16(rs[p]);
      l8[p] = l8[p] * scl[p] + rs[p];
    }
#pragma unroll
    for (int n = 0; n < 8; ++n)
#pragma unroll
      for (int p = 0; p < 8; ++p) acc[n][p] *= scl[p];

    // ---- P·V -------------------------------------------------------------
    v16bf pa = loadA_bf16(&sP[r][0][0], 34);
#pragma unroll
    for (int n = 0; n < 8; ++n) {
      v16bf bv = loadB_rm_f32(&Vt[0][n * 16], KVPITCH);
      acc[n] = wmma_bf16(pa, bv, acc[n]);
    }
    __syncthreads();  // all waves done reading cur; its buffer may be refilled
    cur = nxt;
  }

  // ---- emit split partials ----------------------------------------------
  const size_t pbase = ((((size_t)split * BATCH + b) * HKV + g) * NREP + r) * TQ;
#pragma unroll
  for (int n = 0; n < 8; ++n)
#pragma unroll
    for (int p = 0; p < 8; ++p)
      y_part[(pbase + mrow + p) * HD + n * 16 + cn] = acc[n][p];
  if (cn == 0) {
#pragma unroll
    for (int p = 0; p < 8; ++p) {
      m_part[pbase + mrow + p] = m8[p];
      l_part[pbase + mrow + p] = l8[p];
    }
  }
}

// ---------------------------------------------------------------------------
// Kernel 4: combine split partials (log-sum-exp) -> y_ws [256][4096]
// ---------------------------------------------------------------------------
__global__ __launch_bounds__(128) void reduce_kernel(
    const float* __restrict__ y_part, const float* __restrict__ m_part,
    const float* __restrict__ l_part, float* __restrict__ y_ws) {
  const int bi = blockIdx.x;  // 0 .. B*HQ*TQ-1
  const int t  = bi & 15;
  const int hh = (bi >> 4) & 31;
  const int b  = bi >> 9;
  const int g  = hh >> 2, r = hh & 3;
  const int d  = threadIdx.x;  // 0..127

  float mj[NSPLIT], lj[NSPLIT], mstar = -1e30f;
#pragma unroll
  for (int j = 0; j < NSPLIT; ++j) {
    const size_t idx = ((((size_t)j * BATCH + b) * HKV + g) * NREP + r) * TQ + t;
    mj[j] = m_part[idx];
    lj[j] = l_part[idx];
    mstar = fmaxf(mstar, mj[j]);
  }
  float acc = 0.0f, L = 0.0f;
#pragma unroll
  for (int j = 0; j < NSPLIT; ++j) {
    const float w = __expf(mj[j] - mstar);
    const size_t idx = ((((size_t)j * BATCH + b) * HKV + g) * NREP + r) * TQ + t;
    acc += y_part[idx * HD + d] * w;
    L   += lj[j] * w;
  }
  y_ws[((size_t)(b * TQ + t)) * DIMW + hh * HD + d] = acc / L;
}

// ---------------------------------------------------------------------------
// Host launcher
// ---------------------------------------------------------------------------
extern "C" void kernel_launch(void* const* d_in, const int* in_sizes, int n_in,
                              void* d_out, int out_size, void* d_ws,
                              size_t ws_size, hipStream_t stream) {
  (void)in_sizes; (void)n_in; (void)out_size; (void)ws_size;
  const float* x  = (const float*)d_in[0];
  const int*   ip = (const int*)d_in[3];
  const float* kc = (const float*)d_in[5];
  const float* vc = (const float*)d_in[6];
  const float* wq = (const float*)d_in[7];
  const float* wk = (const float*)d_in[8];
  const float* wv = (const float*)d_in[9];
  const float* wo = (const float*)d_in[10];
  float* out = (float*)d_out;

  // workspace layout (floats)
  float*  ws     = (float*)d_ws;
  float*  qraw   = ws;                                    // 256*4096 (reused as y_ws)
  float*  kraw   = qraw + (size_t)256 * 4096;             // 256*1024
  float*  vraw   = kraw + (size_t)256 * 1024;             // 256*1024
  __bf16* q_att  = (__bf16*)(vraw + (size_t)256 * 1024);  // 256*4096 bf16
  float*  k_new  = (float*)((char*)q_att + (size_t)256 * 4096 * 2);
  float*  v_new  = k_new + (size_t)BATCH * HKV * TQ * HD;
  float*  y_part = v_new + (size_t)BATCH * HKV * TQ * HD;
  float*  m_part = y_part + (size_t)NSPLIT * BATCH * HKV * NREP * TQ * HD;
  float*  l_part = m_part + (size_t)NSPLIT * BATCH * HKV * NREP * TQ;
  float*  y_ws   = qraw;  // alias: qraw dead after rope_scatter

  // 1) projections
  gemm_xwT<<<dim3(4096 / 128, 2), 256, 0, stream>>>(x, wq, qraw, 256, 4096, 4096);
  gemm_xwT<<<dim3(1024 / 128, 2), 256, 0, stream>>>(x, wk, kraw, 256, 1024, 4096);
  gemm_xwT<<<dim3(1024 / 128, 2), 256, 0, stream>>>(x, wv, vraw, 256, 1024, 4096);
  // 2) RoPE + layout scatter
  {
    const int total = 256 * 32 * 64 + 2 * 256 * 8 * 64;
    rope_scatter<<<(total + 255) / 256, 256, 0, stream>>>(qraw, kraw, vraw, ip,
                                                          q_att, k_new, v_new);
  }
  // 3) flash-decode attention over S splits
  attn_kernel<<<dim3(BATCH, HKV, NSPLIT), 128, 0, stream>>>(
      kc, vc, q_att, k_new, v_new, y_part, m_part, l_part);
  // 4) combine splits
  reduce_kernel<<<BATCH * HQ * TQ, 128, 0, stream>>>(y_part, m_part, l_part, y_ws);
  // 5) output projection
  gemm_xwT<<<dim3(4096 / 128, 2), 256, 0, stream>>>(y_ws, wo, out, 256, 4096, 4096);
}
